// SiameseNetwork_41979010351728
// MI455X (gfx1250) — matrix operand order repaired and checked by reading, not compiled, not run on hardware
//
#include <hip/hip_runtime.h>
#include <math.h>

typedef __attribute__((ext_vector_type(16))) _Float16 v16h;
typedef __attribute__((ext_vector_type(8)))  float    v8f;
typedef __attribute__((ext_vector_type(2)))  _Float16 h2;

union FH16 { v16h v; _Float16 e[16]; unsigned u[8]; };
union VF8  { v8f  v; float    e[8];  };
union H2U  { h2   h; unsigned u;     };

#define N_S   1024
#define N_Q   1024
#define DIM   128
#define HID   64
#define NWAY  8
#define QBLK  4      // queries per block
#define SBLK  256    // supports per block (16 tiles of 16)

// Pack two f32 into packed f16 (v_cvt_pk_f16_f32).
__device__ __forceinline__ h2 pk2(float x, float y)
{
    h2 r; r.x = (_Float16)x; r.y = (_Float16)y; return r;
}

// |a - b| on packed f16: v_pk_add_f16(neg) + integer sign-mask.
// (Integer abs avoids the gfx1250 isel crash on vector-f32 fabs patterns.)
__device__ __forceinline__ unsigned absdiff_h2(h2 a, h2 b)
{
    H2U d; d.h = a - b;
    return d.u & 0x7FFF7FFFu;
}

// Grid: 1024 blocks = 256 query-chunks (4 queries) x 4 support-chunks (256 supports).
// 8 waves/block; wave w owns s-tiles {w, w+8} of its chunk. Per s-tile the wave
// caches the 16-support slice as packed f16 in registers, then loops the 4 queries:
//   WMMA M = 16 supports, N = 64 hidden (4 tiles), K = 128 dim (4 steps of 32).
// W1^T (B fragments) is f16 register-resident for the whole kernel.
__global__ void __launch_bounds__(256)
siamese_sim_kernel(const float* __restrict__ sx, const int* __restrict__ sy,
                   const float* __restrict__ qx, const float* __restrict__ W1,
                   const float* __restrict__ b1, const float* __restrict__ W2,
                   float* __restrict__ sums /* [N_Q][NWAY], pre-zeroed */)
{
    __shared__ float cls_sum[QBLK][NWAY];
    const int tid  = threadIdx.x;
    const int lane = tid & 31;
    const int wave = tid >> 5;   // 0..7
    const int m    = lane & 15;  // row-within-tile / column select
    const int hs   = lane >> 4;  // half-wave select
    const int qc   = blockIdx.x >> 2;        // query chunk: q0 = qc*4
    const int sc   = blockIdx.x & 3;         // support chunk: s in [sc*256, sc*256+256)
    const int q0   = qc * QBLK;

    if (tid < QBLK * NWAY) cls_sum[tid >> 3][tid & 7] = 0.0f;
    __syncthreads();

    // ---- B fragments: W1^T, f32 -> f16, register-resident ----
    // B layout (16-bit, 32x16): lane n = column N (n = lane&15); halves 0..7 -> K = 8*hs+i, halves 8..15 -> K = 16+8*hs+i.
    v16h Bf[4][4];
#pragma unroll
    for (int k = 0; k < 4; ++k) {
#pragma unroll
        for (int t = 0; t < 4; ++t) {
            const float* w1p = W1 + (size_t)(16 * t + m) * DIM + 32 * k + hs * 8;
            float4 f0 = *(const float4*)(w1p + 0);
            float4 f1 = *(const float4*)(w1p + 4);
            float4 f2 = *(const float4*)(w1p + 16);
            float4 f3 = *(const float4*)(w1p + 20);
            FH16 b;
            H2U p;
            p.h = pk2(f0.x, f0.y); b.u[0] = p.u;
            p.h = pk2(f0.z, f0.w); b.u[1] = p.u;
            p.h = pk2(f1.x, f1.y); b.u[2] = p.u;
            p.h = pk2(f1.z, f1.w); b.u[3] = p.u;
            p.h = pk2(f2.x, f2.y); b.u[4] = p.u;
            p.h = pk2(f2.z, f2.w); b.u[5] = p.u;
            p.h = pk2(f3.x, f3.y); b.u[6] = p.u;
            p.h = pk2(f3.z, f3.w); b.u[7] = p.u;
            Bf[k][t] = b.v;
        }
    }

    // Per-lane epilogue constants: hidden index per N-tile is 16*t + m.
    float b1v[4], w2v[4];
#pragma unroll
    for (int t = 0; t < 4; ++t) {
        const int h = 16 * t + m;
        b1v[t] = b1[h];
        w2v[t] = W2[h];
    }

    for (int i = 0; i < 2; ++i) {
        const int s0 = (sc * 16 + wave + 8 * i) * 16;   // 16-support tile base
        const float* srow = sx + (size_t)(s0 + m) * DIM;

        // Cache this lane's support slice as packed f16 (reused by all QBLK queries).
        // A layout (16-bit, 16x32): halves 0..7 -> K = kb+i, halves 8..15 -> K = kb+16+i,
        // with kb = 32*k + hs*8.
        h2 sup[4][8];
#pragma unroll
        for (int k = 0; k < 4; ++k) {
            const int kb = 32 * k + hs * 8;
            float4 sa = *(const float4*)(srow + kb + 0);
            float4 sb = *(const float4*)(srow + kb + 4);
            float4 sc4 = *(const float4*)(srow + kb + 16);
            float4 sd = *(const float4*)(srow + kb + 20);
            sup[k][0] = pk2(sa.x, sa.y);
            sup[k][1] = pk2(sa.z, sa.w);
            sup[k][2] = pk2(sb.x, sb.y);
            sup[k][3] = pk2(sb.z, sb.w);
            sup[k][4] = pk2(sc4.x, sc4.y);
            sup[k][5] = pk2(sc4.z, sc4.w);
            sup[k][6] = pk2(sd.x, sd.y);
            sup[k][7] = pk2(sd.z, sd.w);
        }

        for (int qi = 0; qi < QBLK; ++qi) {
            const float* qrow = qx + (size_t)(q0 + qi) * DIM;

            v8f acc0 = {}, acc1 = {}, acc2 = {}, acc3 = {};
#pragma unroll
            for (int k = 0; k < 4; ++k) {
                const int kb = 32 * k + hs * 8;
                float4 qa = *(const float4*)(qrow + kb + 0);
                float4 qb = *(const float4*)(qrow + kb + 4);
                float4 qc4 = *(const float4*)(qrow + kb + 16);
                float4 qd = *(const float4*)(qrow + kb + 20);
                FH16 a;
                a.u[0] = absdiff_h2(pk2(qa.x, qa.y), sup[k][0]);
                a.u[1] = absdiff_h2(pk2(qa.z, qa.w), sup[k][1]);
                a.u[2] = absdiff_h2(pk2(qb.x, qb.y), sup[k][2]);
                a.u[3] = absdiff_h2(pk2(qb.z, qb.w), sup[k][3]);
                a.u[4] = absdiff_h2(pk2(qc4.x, qc4.y), sup[k][4]);
                a.u[5] = absdiff_h2(pk2(qc4.z, qc4.w), sup[k][5]);
                a.u[6] = absdiff_h2(pk2(qd.x, qd.y), sup[k][6]);
                a.u[7] = absdiff_h2(pk2(qd.z, qd.w), sup[k][7]);

                acc0 = __builtin_amdgcn_wmma_f32_16x16x32_f16(false, a.v, false, Bf[k][0], (short)0, acc0, false, false);
                acc1 = __builtin_amdgcn_wmma_f32_16x16x32_f16(false, a.v, false, Bf[k][1], (short)0, acc1, false, false);
                acc2 = __builtin_amdgcn_wmma_f32_16x16x32_f16(false, a.v, false, Bf[k][2], (short)0, acc2, false, false);
                acc3 = __builtin_amdgcn_wmma_f32_16x16x32_f16(false, a.v, false, Bf[k][3], (short)0, acc3, false, false);
            }

            // Epilogue: C layout -> VGPR r holds M = r + 8*hs, lane&15 = N.
            // sim[pair] = sum_H relu(h + b1) * W2  (b2 added in finalize kernel).
            VF8 c0, c1, c2, c3;
            c0.v = acc0; c1.v = acc1; c2.v = acc2; c3.v = acc3;
#pragma unroll
            for (int r = 0; r < 8; ++r) {
                float v0 = c0.e[r] + b1v[0]; v0 = v0 > 0.0f ? v0 : 0.0f;
                float v1 = c1.e[r] + b1v[1]; v1 = v1 > 0.0f ? v1 : 0.0f;
                float v2 = c2.e[r] + b1v[2]; v2 = v2 > 0.0f ? v2 : 0.0f;
                float v3 = c3.e[r] + b1v[3]; v3 = v3 > 0.0f ? v3 : 0.0f;
                float partial = v0 * w2v[0];
                partial = fmaf(v1, w2v[1], partial);
                partial = fmaf(v2, w2v[2], partial);
                partial = fmaf(v3, w2v[3], partial);
                // reduce the 16 lanes of each half-wave (H-dimension reduction)
                partial += __shfl_xor(partial, 1, 32);
                partial += __shfl_xor(partial, 2, 32);
                partial += __shfl_xor(partial, 4, 32);
                partial += __shfl_xor(partial, 8, 32);
                if (m == 0) {
                    const int s = s0 + r + hs * 8;
                    const int cls = sy[s] & (NWAY - 1);
                    atomicAdd(&cls_sum[qi][cls], partial);
                }
            }
        }
    }

    __syncthreads();
    if (tid < QBLK * NWAY) {
        const int qi = tid >> 3, c = tid & 7;
        atomicAdd(&sums[(size_t)(q0 + qi) * NWAY + c], cls_sum[qi][c]);
    }
}

// Finalize: per-class mean + b2, then log_softmax. 4 blocks x 256 threads = 1024 queries.
__global__ void __launch_bounds__(256)
siamese_final_kernel(const float* __restrict__ sums, const int* __restrict__ sy,
                     const float* __restrict__ b2, float* __restrict__ out)
{
    __shared__ float cnt[NWAY];
    const int tid = threadIdx.x;
    if (tid < NWAY) cnt[tid] = 0.0f;
    __syncthreads();
    for (int i = tid; i < N_S; i += 256)
        atomicAdd(&cnt[sy[i] & (NWAY - 1)], 1.0f);
    __syncthreads();

    const int q = blockIdx.x * 256 + tid;
    if (q >= N_Q) return;
    const float bias = b2[0];

    float logit[NWAY];
    float mx = -INFINITY;
#pragma unroll
    for (int c = 0; c < NWAY; ++c) {
        logit[c] = sums[(size_t)q * NWAY + c] / cnt[c] + bias;
        mx = fmaxf(mx, logit[c]);
    }
    float se = 0.0f;
#pragma unroll
    for (int c = 0; c < NWAY; ++c) se += expf(logit[c] - mx);
    const float lse = logf(se);
#pragma unroll
    for (int c = 0; c < NWAY; ++c) out[(size_t)q * NWAY + c] = logit[c] - mx - lse;
}

extern "C" void kernel_launch(void* const* d_in, const int* in_sizes, int n_in,
                              void* d_out, int out_size, void* d_ws, size_t ws_size,
                              hipStream_t stream)
{
    const float* sx = (const float*)d_in[0];   // support_x (1024,128)
    const int*   sy = (const int*)  d_in[1];   // support_y (1024,)
    const float* qx = (const float*)d_in[2];   // query_x   (1024,128)
    // d_in[3] = n_way scalar (8), baked in
    const float* W1 = (const float*)d_in[4];   // (64,128)
    const float* b1 = (const float*)d_in[5];   // (64,)
    const float* W2 = (const float*)d_in[6];   // (1,64)
    const float* b2 = (const float*)d_in[7];   // (1,)
    float* out  = (float*)d_out;               // (1024,8)
    float* sums = (float*)d_ws;                // (1024,8) accumulators

    hipMemsetAsync(sums, 0, (size_t)N_Q * NWAY * sizeof(float), stream);
    siamese_sim_kernel<<<N_Q, 256, 0, stream>>>(sx, sy, qx, W1, b1, W2, sums);
    siamese_final_kernel<<<(N_Q + 255) / 256, 256, 0, stream>>>(sums, sy, b2, out);
}
